// NeuralConvSelection_13460427505897
// MI455X (gfx1250) — matrix operand order repaired and checked by reading, not compiled, not run on hardware
//
#include <hip/hip_runtime.h>
#include <hip/hip_bf16.h>

typedef __bf16 v16bf __attribute__((ext_vector_type(16)));
typedef float  v8f   __attribute__((ext_vector_type(8)));
typedef unsigned int v4u __attribute__((ext_vector_type(4)));
typedef int    v8i   __attribute__((ext_vector_type(8)));
typedef int    v4i   __attribute__((ext_vector_type(4)));

#define NB    8
#define CH    256
#define HWD   96
#define KSZ   8
#define STR   3
#define VC    512
#define OHW   30          // (96-8)/3+1
#define LPP   (OHW*OHW)   // 900
#define NP    (NB*LPP)    // 7200
#define INDIM 320
#define HID   768

// packed-B element counts
#define E_CW  (8*32*512)    // K=256 -> 8 ktiles, N=512 -> 32 ntiles
#define E_W1  (10*48*512)   // K=320, N=768
#define E_W2  (24*16*512)   // K=768, N=256

static __device__ __forceinline__ __bf16 f2bf(float f) { return (__bf16)f; }

// ---------------------------------------------------------------------------
// Tensor Data Mover: build D# groups per CDNA5 ISA 8.3-8.6 and issue
// tensor_load_to_lds. 3D tile (x=tile0, y=tile1, z=tile2), f32 elements.
//   addr = global_addr + 4*(x + y*s0 + z*s1); OOB reads (vs tdimN) return 0.
// ---------------------------------------------------------------------------
static __device__ __forceinline__ void tdm_load_f32(
    const void* gptr, unsigned lds_off,
    unsigned tile0, unsigned tile1, unsigned tile2,
    unsigned tdim0, unsigned tdim1, unsigned tdim2,
    unsigned long long s0, unsigned long long s1) {
    unsigned long long ga = (unsigned long long)(uintptr_t)gptr;
    v4u g0;
    g0[0] = 1u;                                       // count=1, user desc
    g0[1] = lds_off;                                  // lds_addr (bytes)
    g0[2] = (unsigned)ga;                             // global_addr[31:0]
    g0[3] = (unsigned)((ga >> 32) & 0x01FFFFFFu) | (2u << 30);  // addr[56:32] | type=2
    v8i g1;
    g1[0] = (int)(2u << 16);                          // data_size=4B; no mask/flags
    g1[1] = (int)((tdim0 & 0xFFFFu) << 16);           // barrier_addr=0 | tdim0.lo16
    g1[2] = (int)((tdim0 >> 16) | ((tdim1 & 0xFFFFu) << 16));
    g1[3] = (int)((tdim1 >> 16) | (tile0 << 16));
    g1[4] = (int)(tile1 | (tile2 << 16));
    g1[5] = (int)(s0 & 0xFFFFFFFFu);                  // dim0_stride[31:0]
    g1[6] = (int)(((s0 >> 32) & 0xFFFFu) | ((s1 & 0xFFFFu) << 16));
    g1[7] = (int)((s1 >> 16) & 0xFFFFFFFFu);          // dim1_stride[47:16]
    v4i g2;
    g2[0] = (int)tdim2;                               // tensor_dim2
    g2[1] = 1;                                        // tensor_dim3
    g2[2] = 0;                                        // tensor_dim2_stride lo
    g2[3] = 0;                                        // stride hi | tile_dim3=0
    v4i g3 = {0, 0, 0, 0};
#if __clang_major__ >= 23
    v8i g4 = {0, 0, 0, 0, 0, 0, 0, 0};
    __builtin_amdgcn_tensor_load_to_lds(g0, g1, g2, g3, g4, 0);
#else
    __builtin_amdgcn_tensor_load_to_lds(g0, g1, g2, g3, 0);
#endif
}

// B-fragment mapping (V_WMMA_*_16X16X32 B is 32x16):
//   lane l: n = nt*16 + (l&15); halfword h -> k = 16*(h>>3) + 8*(l>>4) + (h&7)
static __device__ __forceinline__ void pack_elem(const float* src, __bf16* dst,
                                                 int e, int NT, int K) {
    int kt   = e / (NT * 512);
    int r    = e % (NT * 512);
    int nt   = r / 512;
    int r2   = r % 512;
    int lane = r2 >> 4;
    int h    = r2 & 15;
    int n    = nt * 16 + (lane & 15);
    int k    = kt * 32 + ((h >> 3) << 4) + ((lane >> 4) << 3) + (h & 7);
    dst[e]   = f2bf(src[(size_t)n * K + k]);
}

__global__ __launch_bounds__(256)
void pack_kernel(const float* __restrict__ conv_w, const float* __restrict__ W1,
                 const float* __restrict__ W2, __bf16* __restrict__ cw_p,
                 __bf16* __restrict__ w1_p, __bf16* __restrict__ w2_p,
                 float* __restrict__ wn_rcp) {
    int tid = blockIdx.x * blockDim.x + threadIdx.x;
    if (tid < E_CW) {
        pack_elem(conv_w, cw_p, tid, 32, 256);
    } else if (tid < E_CW + E_W1) {
        pack_elem(W1, w1_p, tid - E_CW, 48, 320);
    } else if (tid < E_CW + E_W1 + E_W2) {
        pack_elem(W2, w2_p, tid - E_CW - E_W1, 16, 768);
    } else if (tid < E_CW + E_W1 + E_W2 + VC) {
        int v = tid - (E_CW + E_W1 + E_W2);
        float s = 0.f;
        for (int c = 0; c < CH; ++c) {
            float w = conv_w[(size_t)v * CH + c];
            s += w * w;
        }
        wn_rcp[v] = rsqrtf(s);
    }
}

// A-fragment address for element (m, c): mtile=m>>4, kt=c>>5, kk2=c&31,
//   g=(kk2&15)>>3, lane=(m&15)|(g<<4), h=(kk2&7)|((kk2>>4)<<3)
__global__ __launch_bounds__(256)
void patch_kernel(const float* __restrict__ x, const float* __restrict__ conv_w,
                  const __bf16* __restrict__ cw_p, const float* __restrict__ wn_rcp,
                  float* __restrict__ inp) {
    extern __shared__ __align__(32) char smem[];
    float*  xf      = (float*)smem;                       // 64 KB: [c][m] tile (TDM order)
    __bf16* Alds    = (__bf16*)(smem + 65536);            // 32 KB fragments
    float*  partial = (float*)(smem + 65536 + 32768);     // 1 KB
    float*  xn      = partial + 256;                      // 256 B
    float*  redV    = xn + 64;                            // 1 KB
    int*    redI    = (int*)(redV + 256);                 // 1 KB

    const int blk = blockIdx.x;
    const int b   = blk / LPP;
    const int l   = blk % LPP;
    const int r0  = (l / OHW) * STR;
    const int c0  = (l % OHW) * STR;
    const int t   = threadIdx.x;

    // ---- TDM: DMA the 256x8x8 f32 patch tile into LDS (wave 0 issues) ----
    if (t < 32) {
        const float* gp = x + (((size_t)b * CH) * HWD + r0) * HWD + c0;
        tdm_load_f32(gp, (unsigned)(uintptr_t)xf,
                     /*tile*/ KSZ, KSZ, CH,
                     /*tdim*/ HWD, HWD, CH,
                     /*strides*/ HWD, HWD * HWD);
        __builtin_amdgcn_s_wait_tensorcnt(0);
    }
    __syncthreads();

    // ---- LDS f32 -> bf16 A fragments + per-position sumsq ----
    {
        int m  = t & 63;
        int cg = t >> 6;                       // 4 channel groups of 64
        int mtile = m >> 4, mi = m & 15;
        float sq = 0.f;
        for (int cc = 0; cc < 64; ++cc) {
            int c   = cg * 64 + cc;
            float v = xf[c * 64 + m];
            sq += v * v;
            int kt   = c >> 5, kk2 = c & 31;
            int g    = (kk2 & 15) >> 3;
            int lane = mi | (g << 4);
            int h    = (kk2 & 7) | ((kk2 >> 4) << 3);
            Alds[(((mtile * 8 + kt) * 32 + lane) << 4) | h] = f2bf(v);
        }
        partial[t] = sq;                       // t == cg*64 + m
    }
    __syncthreads();
    if (t < 64)
        xn[t] = rsqrtf(partial[t] + partial[64 + t] + partial[128 + t] + partial[192 + t]);
    __syncthreads();

    // ---- 64x512x256 GEMM via WMMA + branchless running argmax ----
    const int wave = t >> 5;
    const int lane = t & 31;
    const int wmt  = wave & 3;    // M tile (0..3)
    const int wng  = wave >> 2;   // N half (0..1)
    const int g    = lane >> 4;

    v16bf afrag[8];
    for (int kt = 0; kt < 8; ++kt)
        afrag[kt] = *(const v16bf*)&Alds[(((wmt * 8 + kt) * 32 + lane) << 4)];

    float xnv[8];
    for (int r = 0; r < 8; ++r) xnv[r] = xn[wmt * 16 + r + (g << 3)];

    float best = -1e30f;
    int bestIdx = 0x7FFFFFFF;
    for (int i = 0; i < 16; ++i) {
        int ntile = wng * 16 + i;
        v8f acc = {};
        for (int kt = 0; kt < 8; ++kt) {
            v16bf bfrag = *(const v16bf*)&cw_p[(((kt * 32 + ntile) * 32 + lane) << 4)];
            acc = __builtin_amdgcn_wmma_f32_16x16x32_bf16(
                false, afrag[kt], false, bfrag, (short)0, acc, false, false);
        }
        int n = ntile * 16 + (lane & 15);
        float wnr = wn_rcp[n];
        for (int r = 0; r < 8; ++r) {
            int mm = wmt * 16 + r + (g << 3);
            float val = acc[r] * xnv[r] * wnr;          // cosine similarity
            int idx = (n << 6) | mm;                    // v*64 + pos flat order
            bool c1 = (val > best) | ((val == best) & (idx < bestIdx));
            best    = c1 ? val : best;                  // v_cndmask, no exec churn
            bestIdx = c1 ? idx : bestIdx;
        }
    }
    redV[t] = best;
    redI[t] = bestIdx;
    __syncthreads();
    for (int s = 128; s > 0; s >>= 1) {
        if (t < s) {
            float ov = redV[t + s];
            int   oi = redI[t + s];
            bool c1 = (ov > redV[t]) | ((ov == redV[t]) & (oi < redI[t]));
            redV[t] = c1 ? ov : redV[t];
            redI[t] = c1 ? oi : redI[t];
        }
        __syncthreads();
    }
    const float ms    = redV[0];
    const int   bi    = redI[0];
    const int   nstar = bi >> 6;
    const int   mstar = bi & 63;

    // ---- sim_map_max: recompute winning column from LDS fragments ----
    if (t < 64) {
        int mm = t, mt2 = mm >> 4, mi2 = mm & 15;
        float dot = 0.f;
        for (int c = 0; c < CH; ++c) {
            int kt = c >> 5, kk2 = c & 31;
            int gg = (kk2 & 15) >> 3;
            int ln = mi2 | (gg << 4);
            int h  = (kk2 & 7) | ((kk2 >> 4) << 3);
            dot += (float)Alds[(((mt2 * 8 + kt) * 32 + ln) << 4) | h] * conv_w[(size_t)nstar * CH + c];
        }
        inp[(size_t)blk * INDIM + 256 + mm] = dot * xn[mm] * wn_rcp[nstar];
    }
    // ---- integ: max_act*ms + conv_w[nstar]*(1-ms), exact f32 from LDS tile ----
    {
        int c = t;
        float xa = xf[c * 64 + mstar];
        float wv = conv_w[(size_t)nstar * CH + c];
        inp[(size_t)blk * INDIM + c] = xa * ms + wv * (1.f - ms);
    }
}

// C = act(A(MxK,f32) @ Bpacked(bf16) + bias); act 0 = leaky(0.2), 1 = tanh
__global__ __launch_bounds__(256)
void mlp_gemm(const float* __restrict__ A, const __bf16* __restrict__ Bp,
              const float* __restrict__ bias, float* __restrict__ C,
              int M, int K, int N, int act) {
    __shared__ __align__(32) float  Xt[64 * 32];      // 8 KB TDM-staged f32 tile
    __shared__ __align__(32) __bf16 Sl[4 * 32 * 16];  // 4 KB bf16 A fragments
    const int t     = threadIdx.x;
    const int row0  = blockIdx.x * 64;
    const int ncol0 = blockIdx.y * 128;
    const int wave  = t >> 5, lane = t & 31;
    const int wmt   = wave & 3, wng = wave >> 2;
    const int g     = lane >> 4;
    const int KT    = K >> 5;
    const int NTt   = N >> 4;

    v8f acc[4];
    for (int i = 0; i < 4; ++i) acc[i] = (v8f){};

    for (int kt = 0; kt < KT; ++kt) {
        // TDM: 64x32 f32 tile; tensor_dim1 = M-row0 zero-fills the M tail
        if (t < 32) {
            tdm_load_f32(A + (size_t)row0 * K + kt * 32, (unsigned)(uintptr_t)Xt,
                         /*tile*/ 32, 64, 0,
                         /*tdim*/ (unsigned)(K - kt * 32), (unsigned)(M - row0), 1,
                         /*strides*/ (unsigned long long)K, 0ull);
            __builtin_amdgcn_s_wait_tensorcnt(0);
        }
        __syncthreads();
        // convert to bf16 fragment layout
        {
            int m = t >> 2;
            int kbase = (t & 3) << 3;
            int mt2 = m >> 4, mi2 = m & 15;
            for (int j = 0; j < 8; ++j) {
                int kk2 = kbase + j;
                int gg = (kk2 & 15) >> 3;
                int ln = mi2 | (gg << 4);
                int h  = (kk2 & 7) | ((kk2 >> 4) << 3);
                Sl[((mt2 * 32 + ln) << 4) | h] = f2bf(Xt[m * 32 + kk2]);
            }
        }
        __syncthreads();
        v16bf af = *(const v16bf*)&Sl[((wmt * 32 + lane) << 4)];
        for (int i = 0; i < 4; ++i) {
            int ntile = (ncol0 >> 4) + wng * 4 + i;
            v16bf bf = *(const v16bf*)&Bp[(((size_t)kt * NTt + ntile) * 32 + lane) << 4];
            acc[i] = __builtin_amdgcn_wmma_f32_16x16x32_bf16(
                false, af, false, bf, (short)0, acc[i], false, false);
        }
        __syncthreads();
    }

    for (int i = 0; i < 4; ++i) {
        int ntile = (ncol0 >> 4) + wng * 4 + i;
        int col = ntile * 16 + (lane & 15);
        float bv = bias[col];
        for (int r = 0; r < 8; ++r) {
            int row = row0 + wmt * 16 + r + (g << 3);
            if (row < M) {
                float v = acc[i][r] + bv;
                v = act ? tanhf(v) : (v > 0.f ? v : 0.2f * v);
                C[(size_t)row * N + col] = v;
            }
        }
    }
}

extern "C" void kernel_launch(void* const* d_in, const int* in_sizes, int n_in,
                              void* d_out, int out_size, void* d_ws, size_t ws_size,
                              hipStream_t stream) {
    const float* x      = (const float*)d_in[0];
    const float* conv_w = (const float*)d_in[1];
    const float* W1     = (const float*)d_in[2];
    const float* b1     = (const float*)d_in[3];
    const float* W2     = (const float*)d_in[4];
    const float* b2     = (const float*)d_in[5];
    float* out = (float*)d_out;

    char* ws = (char*)d_ws;
    size_t off = 0;
    auto carve = [&](size_t bytes) {
        size_t o = off;
        off = (off + bytes + 255) & ~(size_t)255;
        return o;
    };
    __bf16* cw_p   = (__bf16*)(ws + carve((size_t)E_CW * 2));
    __bf16* w1_p   = (__bf16*)(ws + carve((size_t)E_W1 * 2));
    __bf16* w2_p   = (__bf16*)(ws + carve((size_t)E_W2 * 2));
    float*  wn_rcp = (float*) (ws + carve((size_t)VC * 4));
    float*  inp    = (float*) (ws + carve((size_t)NP * INDIM * 4));
    float*  h1     = (float*) (ws + carve((size_t)NP * HID * 4));
    (void)in_sizes; (void)n_in; (void)out_size; (void)ws_size;

    {   // pack weights + weight norms
        int tot = E_CW + E_W1 + E_W2 + VC;
        pack_kernel<<<(tot + 255) / 256, 256, 0, stream>>>(conv_w, W1, W2, cw_p, w1_p, w2_p, wn_rcp);
    }
    // per-patch cosine GEMM + argmax + feature assembly (dynamic LDS ~99.25 KB)
    size_t smem = 65536 + 32768 + 1024 + 256 + 1024 + 1024;
    patch_kernel<<<NP, 256, smem, stream>>>(x, conv_w, cw_p, wn_rcp, inp);
    // MLP layer 1: (7200x320) @ (320x768) + b1, leaky relu
    mlp_gemm<<<dim3((NP + 63) / 64, HID / 128), 256, 0, stream>>>(inp, w1_p, b1, h1, NP, INDIM, HID, 0);
    // MLP layer 2: (7200x768) @ (768x256) + b2, tanh
    mlp_gemm<<<dim3((NP + 63) / 64, CH / 128), 256, 0, stream>>>(h1, w2_p, b2, out, NP, HID, CH, 1);
}
